// MultiLayerQLSTM_83820581748792
// MI455X (gfx1250) — matrix operand order repaired and checked by reading, not compiled, not run on hardware
//
#include <hip/hip_runtime.h>
#include <hip/hip_bf16.h>

// ---------------- problem constants ----------------
#define S_LEN   2048
#define BATCH   32
#define IN_DIM  256
#define H_DIM   512
#define OUT_DIM 256
#define KTOT    768          // IN_DIM + H_DIM
#define NCOLS   2048         // 4 gates * H_DIM
#define NWG     8            // cooperating workgroups (1 per WGP)

typedef _Float16 h16;
typedef __attribute__((ext_vector_type(16))) _Float16 v16h;
typedef __attribute__((ext_vector_type(8)))  float    v8f;

union V16U { int4 q[2]; v16h v; };

// gfx1250 hardware tanh (V_TANH_F32) when the builtin exists; libm fallback.
#if defined(__HIP_DEVICE_COMPILE__) && __has_builtin(__builtin_amdgcn_tanhf)
__device__ __forceinline__ float fast_tanh(float x) { return __builtin_amdgcn_tanhf(x); }
#else
__device__ __forceinline__ float fast_tanh(float x) { return tanhf(x); }
#endif
// sigmoid via tanh: one trans op instead of exp+rcp
__device__ __forceinline__ float fast_sigmoid(float x) {
  return 0.5f + 0.5f * fast_tanh(0.5f * x);
}

// ---- WMMA tile loaders (contiguous k order, same for A and B, wave32) ----
// A (16x32, row-major rows of K): lane m (0..15) holds row m K=[k0..k0+15];
// lane m+16 holds row m K=[k0+16..k0+31].
__device__ __forceinline__ v16h load_tile_A(const h16* rowPtr, int k0, int lh) {
  V16U u;
  const char* p = (const char*)rowPtr + (size_t)((k0 + lh * 16) * 2);
  u.q[0] = *(const int4*)p;
  u.q[1] = *(const int4*)(p + 16);
  return u.v;
}
// B (32x16, stored column-major: column n contiguous over K): lane n holds
// col n K=[k0..k0+15]; lane n+16 holds col n K=[k0+16..k0+31].
__device__ __forceinline__ v16h load_tile_B(const h16* colPtr, int k0, int lh) {
  V16U u;
  const char* p = (const char*)colPtr + (size_t)((k0 + lh * 16) * 2);
  u.q[0] = *(const int4*)p;
  u.q[1] = *(const int4*)(p + 16);
  return u.v;
}

__device__ __forceinline__ v8f wmma_f16(v16h a, v16h b, v8f c) {
  // v_wmma_f32_16x16x32_f16 : D(16x16 f32) = A(16x32 f16) x B(32x16 f16) + C
  return __builtin_amdgcn_wmma_f32_16x16x32_f16(false, a, false, b, (short)0, c,
                                                false, false);
}

// ---- CDNA5 async global->LDS copy (ASYNCcnt-tracked, no VGPR round-trip) ----
__device__ __forceinline__ void async_ld_b128(unsigned lds_off, const void* gptr) {
  asm volatile("global_load_async_to_lds_b128 %0, %1, off"
               :: "v"(lds_off), "v"(gptr) : "memory");
}
__device__ __forceinline__ void wait_async0() {
  asm volatile("s_wait_asynccnt 0" ::: "memory");
}

// ---------------- quaternion Hamilton-product expansion ----------------
__device__ const int   qtab[4][4] = {{0,1,2,3},{1,0,3,2},{2,3,0,1},{3,2,1,0}};
__device__ const float stab[4][4] = {{ 1.f,-1.f,-1.f,-1.f},
                                     { 1.f, 1.f,-1.f, 1.f},
                                     { 1.f, 1.f, 1.f,-1.f},
                                     { 1.f,-1.f, 1.f, 1.f}};

// Combined weight matrix Mt, COLUMN-major: Mt[col*768 + row], col = g*512+c.
// rows 0..255 : W_g (IN x H) from Wx(4,4,64,128); rows 256..767 : U_g from Uh(4,4,128,128)
__global__ void expand_weights(const float* __restrict__ Wx,
                               const float* __restrict__ Uh,
                               h16* __restrict__ Mt) {
  const int col = blockIdx.x;    // 0..2047
  const int row = threadIdx.x;   // 0..767
  const int g  = col >> 9;
  const int c  = col & 511;
  const int bj = c >> 7;
  const int b  = c & 127;
  float v;
  if (row < IN_DIM) {
    const int bi = row >> 6, a = row & 63;               // n4 = 64
    const int q = qtab[bj][bi];
    v = stab[bj][bi] * Wx[((g * 4 + q) * 64 + a) * 128 + b];
  } else {
    const int rr = row - IN_DIM;
    const int bi = rr >> 7, a = rr & 127;                // n4 = 128
    const int q = qtab[bj][bi];
    v = stab[bj][bi] * Uh[((g * 4 + q) * 128 + a) * 128 + b];
  }
  Mt[(size_t)col * KTOT + row] = (h16)v;
}

__global__ void expand_fco(const float* __restrict__ fco_w, h16* __restrict__ fco) {
  const int i = blockIdx.x * 256 + threadIdx.x;
  if (i < OUT_DIM * H_DIM) fco[i] = (h16)fco_w[i];
}

// x (S,B,IN) f32 -> f16 once per launch (takes cvt off the serial path,
// enables pure async b128 staging in the recurrence)
__global__ void convert_x(const float* __restrict__ x, h16* __restrict__ xh) {
  const int i = blockIdx.x * 256 + threadIdx.x;            // float4 index
  const float4 v = ((const float4*)x)[i];
  h16* d = xh + (size_t)i * 4;
  d[0] = (h16)v.x; d[1] = (h16)v.y; d[2] = (h16)v.z; d[3] = (h16)v.w;
}

__global__ void bar_init(unsigned* bar) {
  if (threadIdx.x < 2) bar[threadIdx.x] = 0u;
}

// ---------------- per-step grid barrier (8 resident WGs) ----------------
__device__ __forceinline__ void grid_barrier(unsigned* bar, unsigned target) {
  __threadfence();
  __syncthreads();
  if (threadIdx.x == 0) {
    unsigned prev = __hip_atomic_fetch_add(bar + 0, 1u, __ATOMIC_ACQ_REL,
                                           __HIP_MEMORY_SCOPE_AGENT);
    if (prev == NWG - 1) {
      __hip_atomic_store(bar + 0, 0u, __ATOMIC_RELAXED, __HIP_MEMORY_SCOPE_AGENT);
      __hip_atomic_fetch_add(bar + 1, 1u, __ATOMIC_RELEASE, __HIP_MEMORY_SCOPE_AGENT);
    }
    while (__hip_atomic_load(bar + 1, __ATOMIC_ACQUIRE, __HIP_MEMORY_SCOPE_AGENT) < target)
      __builtin_amdgcn_s_sleep(2);
  }
  __syncthreads();
}

// ---------------- fused output projection: out_t = h_t @ fco^T + b ----------
// Run by waves 0..3; WG owns OUT columns [wgid*32, wgid*32+32).
__device__ __forceinline__ void out_proj(int tOut, int wgid, int wv, int ln, int lh,
                                         const h16 (*zA)[KTOT],
                                         const h16* __restrict__ fco,
                                         const float* __restrict__ fcob,
                                         float* __restrict__ out) {
  const int mt = wv >> 1, nt = wv & 1;
  const int col = wgid * 32 + nt * 16 + ln;
  const h16* arow = &zA[mt * 16 + ln][IN_DIM];        // hidden part of z = h_t
  const h16* colp = fco + (size_t)col * H_DIM;
  v8f acc = {};
  v16h bc = load_tile_B(colp, 0, lh);                 // 2-deep B pipeline
  #pragma unroll 4
  for (int kk = 0; kk < H_DIM; kk += 32) {
    const int kn = (kk + 32 < H_DIM) ? kk + 32 : 0;
    const v16h a  = load_tile_A(arow, kk, lh);
    const v16h bn = load_tile_B(colp, kn, lh);
    acc = wmma_f16(a, bc, acc);
    bc = bn;
  }
  const float bb = fcob[col];
  float* op = out + (size_t)tOut * BATCH * OUT_DIM;
  #pragma unroll
  for (int r = 0; r < 8; ++r) {
    const int row = mt * 16 + lh * 8 + r;             // C/D layout: M = r (+8 hi half)
    op[row * OUT_DIM + col] = acc[r] + bb;
  }
}

// ---------------- persistent recurrence kernel (8 WGs x 256 threads) --------
__global__ void __launch_bounds__(256)
qlstm_recurrence(const h16*  __restrict__ xh,
                 const h16*  __restrict__ Mt,
                 const h16*  __restrict__ fco,
                 const float* __restrict__ fcob,
                 h16*  __restrict__ hbuf,     // [2][32][512] ping-pong
                 unsigned* __restrict__ bar,
                 float* __restrict__ out) {
  __shared__ __align__(16) h16  zA[BATCH][KTOT];      // [x_t | h_{t-1}]  48 KB
  __shared__ float preLds[4][BATCH][64];              // gate pre-acts     32 KB
  __shared__ float cLds[BATCH][64];                   // cell state         8 KB

  const int tid  = threadIdx.x;
  const int wgid = blockIdx.x;                        // 0..7, owns 64 hidden units
  const int wv   = tid >> 5;
  const int lane = tid & 31;
  const int ln   = lane & 15;
  const int lh   = lane >> 4;
  const unsigned zbase = (unsigned)(uintptr_t)&zA[0][0];   // LDS byte offset

  // h_{-1} = 0, c_{-1} = 0
  for (int e = tid; e < BATCH * 64; e += 256) cLds[e >> 6][e & 63] = 0.f;
  for (int e = tid; e < BATCH * H_DIM; e += 256) zA[e >> 9][IN_DIM + (e & 511)] = (h16)0.f;

  for (int t = 0; t < S_LEN; ++t) {
    // ---- async-stage z = [x_t , h_{t-1}] into LDS (no VGPR round-trip) ----
    {
      const char* xsrc = (const char*)(xh + (size_t)t * BATCH * IN_DIM);
      for (int e = tid; e < BATCH * IN_DIM * 2 / 16; e += 256) {   // 1024 x b128
        const int row = e >> 5, c16 = e & 31;                      // 32 chunks/row
        async_ld_b128(zbase + (unsigned)(row * (KTOT * 2) + c16 * 16), xsrc + e * 16);
      }
      if (t > 0) {
        const char* hsrc = (const char*)(hbuf + (size_t)((t & 1) ^ 1) * BATCH * H_DIM);
        for (int e = tid; e < BATCH * H_DIM * 2 / 16; e += 256) {  // 2048 x b128
          const int row = e >> 6, c16 = e & 63;                    // 64 chunks/row
          async_ld_b128(zbase + (unsigned)(row * (KTOT * 2) + IN_DIM * 2 + c16 * 16),
                        hsrc + e * 16);
        }
      }
      wait_async0();
    }
    __syncthreads();

    // ---- overlapped: project h_{t-1} to out[t-1] (waves 0..3) ----
    if (t > 0 && wv < 4) out_proj(t - 1, wgid, wv, ln, lh, zA, fco, fcob, out);

    // ---- gate GEMM: (32x768) @ Mt_sub(768x256) -> pre (4 gates x 32 x 64) ----
    {
      const int mtile = wv >> 2;                      // 0..1 (batch rows 0-15 / 16-31)
      const int gate  = wv & 3;                       // f,i,o,c
      const h16* arow = &zA[mtile * 16 + ln][0];
      const size_t colBase = (size_t)(gate * H_DIM + wgid * 64);
      #pragma unroll
      for (int nt = 0; nt < 4; ++nt) {
        const h16* colp = Mt + (colBase + nt * 16 + ln) * KTOT;
        v8f acc = {};
        v16h bc = load_tile_B(colp, 0, lh);           // 2-deep B pipeline
        #pragma unroll 4
        for (int k0 = 0; k0 < KTOT; k0 += 32) {
          const int kn = (k0 + 32 < KTOT) ? k0 + 32 : 0;
          const v16h a  = load_tile_A(arow, k0, lh);
          const v16h bn = load_tile_B(colp, kn, lh);
          acc = wmma_f16(a, bc, acc);
          bc = bn;
        }
        #pragma unroll
        for (int r = 0; r < 8; ++r)
          preLds[gate][mtile * 16 + lh * 8 + r][nt * 16 + ln] = acc[r];
      }
    }
    __syncthreads();

    // ---- elementwise LSTM cell (hardware v_tanh); publish h_t (f16) ----
    {
      h16* hw = hbuf + (size_t)(t & 1) * BATCH * H_DIM;
      for (int e = tid; e < BATCH * 64; e += 256) {
        const int m = e >> 6, j = e & 63;
        const float pf = preLds[0][m][j], pi = preLds[1][m][j];
        const float po = preLds[2][m][j], pc = preLds[3][m][j];
        const float fg = fast_sigmoid(pf);
        const float ig = fast_sigmoid(pi);
        const float og = fast_sigmoid(po);
        const float cv = ig * fast_tanh(pc) + fg * cLds[m][j];
        cLds[m][j] = cv;
        hw[m * H_DIM + wgid * 64 + j] = (h16)(og * fast_tanh(cv));
      }
    }
    grid_barrier(bar, (unsigned)(t + 1));
  }

  // ---- final step's output projection (h_{S-1} lives in hbuf[1]) ----
  {
    const char* hsrc = (const char*)(hbuf + (size_t)((S_LEN - 1) & 1) * BATCH * H_DIM);
    for (int e = tid; e < BATCH * H_DIM * 2 / 16; e += 256) {
      const int row = e >> 6, c16 = e & 63;
      async_ld_b128(zbase + (unsigned)(row * (KTOT * 2) + IN_DIM * 2 + c16 * 16),
                    hsrc + e * 16);
    }
    wait_async0();
  }
  __syncthreads();
  if (wv < 4) out_proj(S_LEN - 1, wgid, wv, ln, lh, zA, fco, fcob, out);
}

// ---------------- host driver ----------------
extern "C" void kernel_launch(void* const* d_in, const int* in_sizes, int n_in,
                              void* d_out, int out_size, void* d_ws, size_t ws_size,
                              hipStream_t stream) {
  const float* x     = (const float*)d_in[0];   // (2048,32,256)
  const float* Wx    = (const float*)d_in[1];   // (4,4,64,128)
  const float* Uh    = (const float*)d_in[2];   // (4,4,128,128)
  const float* fco_w = (const float*)d_in[3];   // (256,512)
  const float* fco_b = (const float*)d_in[4];   // (256,)
  float* out = (float*)d_out;                   // (2048,32,256)

  // workspace layout: f16 weights (~3.4 MB, L2-resident) + f16 x copy (32 MB)
  h16* Mt   = (h16*)d_ws;                             // 2048*768
  h16* fco  = Mt   + (size_t)NCOLS * KTOT;            // 256*512
  h16* hbuf = fco  + (size_t)OUT_DIM * H_DIM;         // 2*32*512 ping-pong
  h16* xh   = hbuf + (size_t)2 * BATCH * H_DIM;       // 2048*32*256
  unsigned* bar = (unsigned*)(xh + (size_t)S_LEN * BATCH * IN_DIM);

  expand_weights<<<NCOLS, KTOT, 0, stream>>>(Wx, Uh, Mt);
  expand_fco<<<(OUT_DIM * H_DIM + 255) / 256, 256, 0, stream>>>(fco_w, fco);
  convert_x<<<(S_LEN * BATCH * IN_DIM / 4) / 256, 256, 0, stream>>>(x, xh);
  bar_init<<<1, 32, 0, stream>>>(bar);
  qlstm_recurrence<<<NWG, 256, 0, stream>>>(xh, Mt, fco, fco_b, hbuf, bar, out);
}
